// SEDenseNet_85495618994749
// MI455X (gfx1250) — compile-verified, matrix operand order checked by reading, not run on hardware
//
#include <hip/hip_runtime.h>
#include <hip/hip_bf16.h>
#include <math.h>

// ---------------------------------------------------------------------------
// NAT (neighborhood attention) transformer block for MI455X / gfx1250.
// GEMMs: v_wmma_f32_16x16x32_bf16, A tile staged to LDS with
// global_load_async_to_lds_b128 (ASYNCcnt), read back via ds_load_b128,
// B streamed from global (L2).  Attention: flash-style online softmax (VALU).
// ---------------------------------------------------------------------------

typedef __attribute__((ext_vector_type(16))) __bf16 v16bf;
typedef __attribute__((ext_vector_type(8)))  float  v8f;

#define MTOK 8192     // B*H*W = 2*64*64
#define CDIM 128
#define BDIM 2
#define HDIM 64
#define WDIM 64
#define NHEAD 4
#define HD 32
#define KW 7
#define KMAX 512
#define KPAD 8        // 16B row skew to spread LDS banks

// fp32 -> bf16 round-to-nearest-even
__device__ __forceinline__ unsigned short f2bf(float f) {
  unsigned u = __float_as_uint(f);
  u += 0x7FFFu + ((u >> 16) & 1u);
  return (unsigned short)(u >> 16);
}

// --------------------------- weight convert/transpose -----------------------
// in:  (K, N) row-major fp32.   out: (N, K) row-major bf16.
__global__ void nat_wT_bf16(const float* __restrict__ in,
                            unsigned short* __restrict__ out, int K, int N) {
  int t = blockIdx.x * blockDim.x + threadIdx.x;
  if (t >= K * N) return;
  int k = t / N, n = t - k * N;
  out[(size_t)n * K + k] = f2bf(in[t]);
}

// ------------------------------- layernorm ----------------------------------
// One wave32 per token; 4 contiguous floats per lane; bf16 output.
__global__ void nat_ln_bf16(const float* __restrict__ x,
                            const float* __restrict__ g,
                            const float* __restrict__ b,
                            unsigned short* __restrict__ out) {
  int gt   = blockIdx.x * blockDim.x + threadIdx.x;
  int tok  = gt >> 5;
  int lane = gt & 31;
  if (tok >= MTOK) return;
  const float4 v = *(const float4*)(x + (size_t)tok * CDIM + lane * 4);
  float s = v.x + v.y + v.z + v.w;
#pragma unroll
  for (int o = 16; o >= 1; o >>= 1) s += __shfl_xor(s, o, 32);
  float mean = s * (1.0f / CDIM);
  float d0 = v.x - mean, d1 = v.y - mean, d2 = v.z - mean, d3 = v.w - mean;
  float s2 = d0 * d0 + d1 * d1 + d2 * d2 + d3 * d3;
#pragma unroll
  for (int o = 16; o >= 1; o >>= 1) s2 += __shfl_xor(s2, o, 32);
  float r = 1.0f / sqrtf(s2 * (1.0f / CDIM) + 1e-5f);
  int c = lane * 4;
  unsigned short* op = out + (size_t)tok * CDIM + c;
  op[0] = f2bf(d0 * r * g[c + 0] + b[c + 0]);
  op[1] = f2bf(d1 * r * g[c + 1] + b[c + 1]);
  op[2] = f2bf(d2 * r * g[c + 2] + b[c + 2]);
  op[3] = f2bf(d3 * r * g[c + 3] + b[c + 3]);
}

// ------------------------------ WMMA GEMM -----------------------------------
// D(M,N) = A(M,K)bf16 * Bt(N,K)bf16^T  (+bias, optional exact GELU, +resid).
// 8 waves / block; wave w owns the 16x16 tile at (blockIdx.x*16, (8*by+w)*16).
// The 16xK A tile (shared by all 8 waves) is staged into LDS once per block
// via global_load_async_to_lds_b128 (tracked on ASYNCcnt), then read back as
// ds_load_b128.  B streams from global (each wave owns its 16 columns).
// A-operand lane layout (16-bit, 16x32): lane<16 -> row=lane, K 0-7 & 16-23;
// lane>=16 -> row=lane-16, K 8-15 & 24-31.  B uses the same layout of B^T.
__global__ void nat_gemm_wmma(const unsigned short* __restrict__ A,
                              const unsigned short* __restrict__ Bt,
                              const float* __restrict__ bias,
                              const float* __restrict__ resid,
                              float* __restrict__ outF,
                              unsigned short* __restrict__ outB,
                              int M, int N, int K, int act) {
  // Sole LDS object in this kernel -> wave-relative LDS byte offset 0.
  __shared__ unsigned short As[16 * (KMAX + KPAD)];

  const int wid  = threadIdx.x >> 5;
  const int lane = threadIdx.x & 31;
  const int m0 = blockIdx.x << 4;
  const int n0 = ((blockIdx.y << 3) + wid) << 4;
  const int ldk = K + KPAD;

  // Input-only escape: lets the async-asm "memory" clobbers act as potential
  // writers of As (so the LDS reads below survive), WITHOUT laundering the
  // pointer -> reads keep addrspace(3) and lower to ds_load_b128.
  asm volatile("" :: "v"((unsigned short*)As) : "memory");

  // ---- async-stage A tile: 16 rows x K bf16, 16 bytes per lane per step ----
  {
    const int nvec = (16 * K) >> 3;               // # of 16B vectors (mult of 256)
    for (int v0 = threadIdx.x; v0 < nvec; v0 += 256) {
      int row  = v0 / (K >> 3);
      int col8 = (v0 - row * (K >> 3)) << 3;      // ushort index within row
      unsigned long long ga =
          (unsigned long long)(A + (size_t)(m0 + row) * K + col8);
      unsigned ldsoff = (unsigned)((row * ldk + col8) * 2);
      asm volatile("global_load_async_to_lds_b128 %0, %1, off"
                   :: "v"(ldsoff), "v"(ga) : "memory");
    }
    asm volatile("s_wait_asynccnt 0x0" ::: "memory");
    __syncthreads();
  }

  const int lm = lane & 15;
  const int hi = lane >> 4;                       // lane-half selects K sub-block
  const unsigned short* ap = As + lm * ldk + hi * 8;              // LDS (as(3))
  const unsigned short* bp = Bt + (size_t)(n0 + lm) * K + hi * 8; // global
  v8f acc = {0.f, 0.f, 0.f, 0.f, 0.f, 0.f, 0.f, 0.f};
  for (int kk = 0; kk < K; kk += 32) {
    union { uint4 u[2]; v16bf v; } am, bm;
    am.u[0] = *(const uint4*)(ap + kk);           // ds_load_b128
    am.u[1] = *(const uint4*)(ap + kk + 16);
    bm.u[0] = *(const uint4*)(bp + kk);           // global_load_b128
    bm.u[1] = *(const uint4*)(bp + kk + 16);
    if (kk + 32 < K) {                            // gfx1250 global_prefetch_b8
      __builtin_prefetch(bp + kk + 32, 0, 1);
    }
    acc = __builtin_amdgcn_wmma_f32_16x16x32_bf16(
        false, am.v, false, bm.v, (short)0, acc, false, false);
  }
  const int n = n0 + lm;
  const float bi = bias ? bias[n] : 0.0f;
#pragma unroll
  for (int r = 0; r < 8; ++r) {
    int m = m0 + r + hi * 8;                      // C/D layout: VGPR r, lane half
    float v = acc[r] + bi;
    if (act) v = 0.5f * v * (1.0f + erff(v * 0.70710678118654752f)); // exact GELU
    if (resid) v += resid[(size_t)m * N + n];
    if (outF) outF[(size_t)m * N + n] = v;
    if (outB) outB[(size_t)m * N + n] = f2bf(v);
  }
}

// --------------------------- neighborhood attention -------------------------
// One thread per (pixel, head). qkv rows: [q(0..127) | k(128..255) | v(256..383)],
// head h occupies 32 lanes at h*32. Flash-style single-pass online softmax over
// the clamped 7x7 window; output written bf16 for the proj WMMA GEMM.
__global__ void nat_attn(const float* __restrict__ qkv,
                         const float* __restrict__ rpb,
                         unsigned short* __restrict__ O) {
  int t = blockIdx.x * blockDim.x + threadIdx.x;
  if (t >= MTOK * NHEAD) return;
  int p = t >> 2;                 // pixel token
  int h = t & 3;                  // head
  int b = p >> 12;
  int rem = p & 4095;
  int i = rem >> 6;
  int j = rem & 63;

  const float* qp = qkv + (size_t)p * 384 + h * HD;
  float q[HD];
#pragma unroll
  for (int d = 0; d < HD; ++d) q[d] = qp[d] * 0.17677669529663687f; // hd^-0.5

  int si = min(max(i - 3, 0), HDIM - KW);
  int sj = min(max(j - 3, 0), WDIM - KW);

  float mmax = -3.0e38f, ssum = 0.0f;
  float acc[HD];
#pragma unroll
  for (int d = 0; d < HD; ++d) acc[d] = 0.0f;

  for (int a = 0; a < KW; ++a) {
    int ki = si + a;
    int ri = (ki - i + 6) * 13;
    for (int c = 0; c < KW; ++c) {
      int kj = sj + c;
      size_t kt = ((size_t)(b * 4096 + ki * 64 + kj)) * 384 + h * HD;
      const float* kp = qkv + kt + 128;
      const float* vp = qkv + kt + 256;
      float s = rpb[h * 169 + ri + (kj - j + 6)];
#pragma unroll
      for (int d = 0; d < HD; ++d) s += q[d] * kp[d];
      float mn   = fmaxf(mmax, s);
      float corr = expf(mmax - mn);
      float e    = expf(s - mn);
      ssum = ssum * corr + e;
#pragma unroll
      for (int d = 0; d < HD; ++d) acc[d] = acc[d] * corr + e * vp[d];
      mmax = mn;
    }
  }
  float inv = 1.0f / ssum;
  unsigned short* op = O + (size_t)p * CDIM + h * HD;
#pragma unroll
  for (int d = 0; d < HD; ++d) op[d] = f2bf(acc[d] * inv);
}

// ------------------------------- launcher -----------------------------------
// Workspace carve (256B-aligned, ~31.9 MB total)
#define OFF_QKVT 0u            // 384*128 bf16  =   98304
#define OFF_PROJT 98304u       // 128*128 bf16  =   32768
#define OFF_FC1T 131072u       // 512*128 bf16  =  131072
#define OFF_FC2T 262144u       // 128*512 bf16  =  131072
#define OFF_XN   393216u       // 8192*128 bf16 = 2097152
#define OFF_QKV  2490368u      // 8192*384 f32  = 12582912
#define OFF_O    15073280u     // 8192*128 bf16 = 2097152
#define OFF_X1   17170432u     // 8192*128 f32  = 4194304
#define OFF_Y    21364736u     // 8192*128 bf16 = 2097152
#define OFF_H    23461888u     // 8192*512 bf16 = 8388608  (end 31850496)

extern "C" void kernel_launch(void* const* d_in, const int* in_sizes, int n_in,
                              void* d_out, int out_size, void* d_ws, size_t ws_size,
                              hipStream_t stream) {
  const float* x       = (const float*)d_in[0];
  const float* norm1_g = (const float*)d_in[1];
  const float* norm1_b = (const float*)d_in[2];
  const float* qkv_w   = (const float*)d_in[3];
  const float* qkv_b   = (const float*)d_in[4];
  const float* rpb     = (const float*)d_in[5];
  const float* proj_w  = (const float*)d_in[6];
  const float* proj_b  = (const float*)d_in[7];
  const float* norm2_g = (const float*)d_in[8];
  const float* norm2_b = (const float*)d_in[9];
  const float* fc1_w   = (const float*)d_in[10];
  const float* fc1_b   = (const float*)d_in[11];
  const float* fc2_w   = (const float*)d_in[12];
  const float* fc2_b   = (const float*)d_in[13];

  char* ws = (char*)d_ws;
  unsigned short* qkvT  = (unsigned short*)(ws + OFF_QKVT);
  unsigned short* projT = (unsigned short*)(ws + OFF_PROJT);
  unsigned short* fc1T  = (unsigned short*)(ws + OFF_FC1T);
  unsigned short* fc2T  = (unsigned short*)(ws + OFF_FC2T);
  unsigned short* xn    = (unsigned short*)(ws + OFF_XN);
  float*          qkvb  = (float*)         (ws + OFF_QKV);
  unsigned short* Obuf  = (unsigned short*)(ws + OFF_O);
  float*          x1    = (float*)         (ws + OFF_X1);
  unsigned short* ybuf  = (unsigned short*)(ws + OFF_Y);
  unsigned short* hmid  = (unsigned short*)(ws + OFF_H);
  float*          out   = (float*)d_out;

  const dim3 blk(256);

  // 1) weight convert + transpose to bf16 (N,K)
  nat_wT_bf16<<<dim3((128 * 384 + 255) / 256), blk, 0, stream>>>(qkv_w, qkvT, 128, 384);
  nat_wT_bf16<<<dim3((128 * 128 + 255) / 256), blk, 0, stream>>>(proj_w, projT, 128, 128);
  nat_wT_bf16<<<dim3((128 * 512 + 255) / 256), blk, 0, stream>>>(fc1_w, fc1T, 128, 512);
  nat_wT_bf16<<<dim3((512 * 128 + 255) / 256), blk, 0, stream>>>(fc2_w, fc2T, 512, 128);

  // 2) LN1 -> bf16
  nat_ln_bf16<<<dim3(MTOK / 8), blk, 0, stream>>>(x, norm1_g, norm1_b, xn);

  // 3) qkv GEMM: (8192x128) x (128x384) + bias -> fp32
  nat_gemm_wmma<<<dim3(MTOK / 16, 3), blk, 0, stream>>>(
      xn, qkvT, qkv_b, nullptr, qkvb, nullptr, MTOK, 384, 128, 0);

  // 4) neighborhood attention -> bf16
  nat_attn<<<dim3(MTOK * NHEAD / 256), blk, 0, stream>>>(qkvb, rpb, Obuf);

  // 5) proj GEMM + residual: x1 = x + O @ proj_w + proj_b -> fp32
  nat_gemm_wmma<<<dim3(MTOK / 16, 1), blk, 0, stream>>>(
      Obuf, projT, proj_b, x, x1, nullptr, MTOK, 128, 128, 0);

  // 6) LN2 -> bf16
  nat_ln_bf16<<<dim3(MTOK / 8), blk, 0, stream>>>(x1, norm2_g, norm2_b, ybuf);

  // 7) fc1 GEMM + exact GELU -> bf16
  nat_gemm_wmma<<<dim3(MTOK / 16, 4), blk, 0, stream>>>(
      ybuf, fc1T, fc1_b, nullptr, nullptr, hmid, MTOK, 512, 128, 1);

  // 8) fc2 GEMM + residual -> d_out (fp32)
  nat_gemm_wmma<<<dim3(MTOK / 16, 1), blk, 0, stream>>>(
      hmid, fc2T, fc2_b, x1, out, nullptr, MTOK, 128, 512, 0);
}
